// APIQMixer_8186207666483
// MI455X (gfx1250) — matrix operand order repaired and checked by reading, not compiled, not run on hardware
//
#include <hip/hip_runtime.h>
#include <hip/hip_bf16.h>
#include <stdint.h>

// Problem constants (from reference)
#define A_N   10
#define NE_N  11
#define FA    32
#define EDIM  64
#define HDIM  256
#define BT_N  16384          // B*T
#define SROW  672            // A*FA + NE*FE
#define ALLY_OFF   0
#define ENEMY_OFF  320       // A*FA

typedef __attribute__((ext_vector_type(16))) __bf16 v16bf;
typedef __attribute__((ext_vector_type(8)))  float  v8f;

union Frag { v16bf v; unsigned int u[8]; };

__device__ __forceinline__ unsigned short f2bf(float f) {
    unsigned int u = __float_as_uint(f);
    unsigned int r = (u + 0x7FFFu + ((u >> 16) & 1u)) >> 16;   // RNE
    return (unsigned short)r;
}
__device__ __forceinline__ unsigned int pack2(float lo, float hi) {
    return (unsigned int)f2bf(lo) | ((unsigned int)f2bf(hi) << 16);
}
__device__ __forceinline__ void atomAdd(float* p, float v) {
    (void)__hip_atomic_fetch_add(p, v, __ATOMIC_RELAXED, __HIP_MEMORY_SCOPE_AGENT);
}

// ---------------------------------------------------------------------------
// Zero the accumulator region of the workspace
// ---------------------------------------------------------------------------
__global__ void zero_acc(float* __restrict__ p, int n) {
    int i = blockIdx.x * blockDim.x + threadIdx.x;
    int stride = gridDim.x * blockDim.x;
    for (; i < n; i += stride) p[i] = 0.f;
}

// ---------------------------------------------------------------------------
// Pre-permute a KxN fp32 weight matrix into bf16 B-fragment blocks.
// B-fragment layout (v_wmma_*_16x16x32_bf16, wave32):
//   lane  = N = nt*16 + (lane&15), h = lane>>4
//   dword v (0..7) = packed (K = (v>>2)*16 + h*8 + (v&3)*2, +1) within the 32-K tile
// Block order: [(nt*KT + kt)*32 + lane]*8 + v   -> each fragment is 2x b128.
// ---------------------------------------------------------------------------
__global__ void pack_w(const float* __restrict__ src, unsigned int* __restrict__ dst,
                       int N, int KT, int NT) {
    int total = KT * NT * 256;
    int i = blockIdx.x * blockDim.x + threadIdx.x;
    int stride = gridDim.x * blockDim.x;
    for (; i < total; i += stride) {
        int v    = i & 7;
        int lane = (i >> 3) & 31;
        int t    = i >> 8;
        int kt   = t % KT;
        int nt   = t / KT;
        int k    = kt * 32 + ((v >> 2) * 16) + (lane >> 4) * 8 + (v & 3) * 2;
        int n0   = nt * 16 + (lane & 15);
        float f0 = (n0 < N) ? src[k * N + n0]       : 0.f;
        float f1 = (n0 < N) ? src[(k + 1) * N + n0] : 0.f;
        dst[i] = pack2(f0, f1);
    }
}

// ---------------------------------------------------------------------------
// Fused per-entity double-MLP + mixer-contribution kernel.
// Block = 128 threads (4 waves), 64 entity rows (16 rows / wave).
// Layer 1: H[64][512] = relu(X @ [W1_mlp1 | W1_mlp2] + b1)  (bf16 in LDS)
// Layer 2: y1 = H[:, :256] @ W2_1 + b2_1   (128 cols: [w | b])
//          y2 = H[:,256:]  @ W2_2 + b2_2   (65 cols, padded to 80)
// Contributions accumulated with fp32 global atomics per (bt, e).
// ---------------------------------------------------------------------------
__global__ __launch_bounds__(128)
void entity_mlp(const float* __restrict__ states,
                const float* __restrict__ qvals,
                const unsigned int* __restrict__ pW1_1, const float* __restrict__ b1_1,
                const unsigned int* __restrict__ pW2_1, const float* __restrict__ b2_1,
                const unsigned int* __restrict__ pW1_2, const float* __restrict__ b1_2,
                const unsigned int* __restrict__ pW2_2, const float* __restrict__ b2_2,
                float* __restrict__ dstW,   // y1 cols [0,64)
                float* __restrict__ dstB,   // y1 cols [64,128)
                float* __restrict__ dstW2,  // y2 cols [0,64)
                float* __restrict__ dstS2,  // y2 col 64 (scalar per bt)
                int nEnt, int featOff, int useQv)
{
    __shared__ unsigned short Hs[64 * 512];          // 64 KB hidden activations (bf16)
    const int tid  = threadIdx.x;
    const int wave = tid >> 5;
    const int lane = tid & 31;
    const int lm   = lane & 15;
    const int kh   = lane >> 4;
    const int waveRowBase = wave * 16;
    const int blockRow0   = blockIdx.x * 64;

    // ---- A fragment, layer 1: 16x32 bf16 from states (fp32 -> bf16) ----
    // lane: M = lm, VGPR v: K = (v>>2)*16 + kh*8 + (v&3)*2
    {
        const int rowA = blockRow0 + waveRowBase + lm;
        const int bta  = rowA / nEnt;
        const int ea   = rowA % nEnt;
        const float* xrow = states + bta * SROW + featOff + ea * FA;
        Frag afrag;
        #pragma unroll
        for (int v = 0; v < 8; ++v) {
            int k = ((v >> 2) * 16) + kh * 8 + (v & 3) * 2;
            afrag.u[v] = pack2(xrow[k], xrow[k + 1]);
        }

        // ---- layer 1: 32 N-tiles (cols 0..255 = MLP1, 256..511 = MLP2) ----
        for (int nt = 0; nt < 32; ++nt) {
            const unsigned int* bp = (nt < 16) ? (pW1_1 + nt * 256 + lane * 8)
                                               : (pW1_2 + (nt - 16) * 256 + lane * 8);
            Frag bfrag;
            #pragma unroll
            for (int v = 0; v < 8; ++v) bfrag.u[v] = bp[v];
            v8f c = {};
            c = __builtin_amdgcn_wmma_f32_16x16x32_bf16(false, afrag.v, false, bfrag.v,
                                                        (short)0, c, false, false);
            const int ncol = (nt & 15) * 16 + lm;
            const float bias = (nt < 16) ? b1_1[ncol] : b1_2[ncol];
            const int hcol = nt * 16 + lm;
            #pragma unroll
            for (int r = 0; r < 8; ++r) {
                int m = kh * 8 + r;                         // C layout: M = kh*8 + r, N = lm
                float hval = fmaxf(c[r] + bias, 0.f);       // +bias, ReLU
                Hs[(waveRowBase + m) * 512 + hcol] = f2bf(hval);
            }
        }
    }
    __syncthreads();

    const unsigned int* HsU = (const unsigned int*)Hs;      // aligned bf16 pairs
    const int rowL = waveRowBase + lm;                      // A-frag row, layer 2

    // ---- layer 2, MLP1: y1 = H[:, :256] @ W2_1 + b2_1  (N=128, 8 tiles) ----
    for (int nt = 0; nt < 8; ++nt) {
        v8f c = {};
        #pragma unroll 2
        for (int kt = 0; kt < 8; ++kt) {
            Frag af;
            #pragma unroll
            for (int v = 0; v < 8; ++v) {
                int k = kt * 32 + ((v >> 2) * 16) + kh * 8 + (v & 3) * 2;
                af.u[v] = HsU[rowL * 256 + (k >> 1)];
            }
            Frag bf_;
            const unsigned int* bp = pW2_1 + ((nt * 8 + kt) * 32 + lane) * 8;
            #pragma unroll
            for (int v = 0; v < 8; ++v) bf_.u[v] = bp[v];
            c = __builtin_amdgcn_wmma_f32_16x16x32_bf16(false, af.v, false, bf_.v,
                                                        (short)0, c, false, false);
        }
        const int n = nt * 16 + lm;
        const float bias = b2_1[n];
        #pragma unroll
        for (int r = 0; r < 8; ++r) {
            int row = blockRow0 + waveRowBase + kh * 8 + r;
            int bt  = row / nEnt;
            int e   = row % nEnt;
            float y = c[r] + bias;
            if (n < EDIM) {
                float contrib = useQv ? (qvals[bt * A_N + e] * fabsf(y)) : y;
                atomAdd(&dstW[bt * EDIM + n], contrib);
            } else {
                atomAdd(&dstB[bt * EDIM + (n - EDIM)], y);
            }
        }
    }

    // ---- layer 2, MLP2: y2 = H[:, 256:] @ W2_2 + b2_2  (N=65 -> 5 tiles) ----
    for (int nt = 0; nt < 5; ++nt) {
        v8f c = {};
        #pragma unroll 2
        for (int kt = 0; kt < 8; ++kt) {
            Frag af;
            #pragma unroll
            for (int v = 0; v < 8; ++v) {
                int k = kt * 32 + ((v >> 2) * 16) + kh * 8 + (v & 3) * 2;
                af.u[v] = HsU[rowL * 256 + ((256 + k) >> 1)];
            }
            Frag bf_;
            const unsigned int* bp = pW2_2 + ((nt * 8 + kt) * 32 + lane) * 8;
            #pragma unroll
            for (int v = 0; v < 8; ++v) bf_.u[v] = bp[v];
            c = __builtin_amdgcn_wmma_f32_16x16x32_bf16(false, af.v, false, bf_.v,
                                                        (short)0, c, false, false);
        }
        const int n = nt * 16 + lm;
        const float bias = (n <= EDIM) ? b2_2[n] : 0.f;
        #pragma unroll
        for (int r = 0; r < 8; ++r) {
            int row = blockRow0 + waveRowBase + kh * 8 + r;
            int bt  = row / nEnt;
            float y = c[r] + bias;
            if (n < EDIM)       atomAdd(&dstW2[bt * EDIM + n], y);
            else if (n == EDIM) atomAdd(&dstS2[bt], y);
        }
    }
}

// ---------------------------------------------------------------------------
// Final mixer: per-bt ELU + weighted reduction (one thread per bt row).
// ---------------------------------------------------------------------------
__global__ void mixer(const float* __restrict__ qvals,
                      const float* __restrict__ accA_hid,
                      const float* __restrict__ accE_w,
                      const float* __restrict__ accE_b,
                      const float* __restrict__ accA_w2,
                      const float* __restrict__ accE_w2,
                      const float* __restrict__ accA_b2,
                      const float* __restrict__ accE_b2,
                      float* __restrict__ out)
{
    int bt = blockIdx.x * blockDim.x + threadIdx.x;
    if (bt >= BT_N) return;
    float sq = 0.f;
    #pragma unroll
    for (int a = 0; a < A_N; ++a) sq += qvals[bt * A_N + a];
    float acc = 0.f;
    for (int e = 0; e < EDIM; ++e) {
        float hid = accA_hid[bt * EDIM + e]
                  + sq * fabsf(accE_w[bt * EDIM + e])
                  + accE_b[bt * EDIM + e];
        float hidden = (hid > 0.f) ? hid : expm1f(hid);      // ELU, alpha=1
        acc += hidden * (fabsf(accA_w2[bt * EDIM + e]) + fabsf(accE_w2[bt * EDIM + e]));
    }
    out[bt] = acc + accA_b2[bt] + accE_b2[bt];
}

// ---------------------------------------------------------------------------
extern "C" void kernel_launch(void* const* d_in, const int* in_sizes, int n_in,
                              void* d_out, int out_size, void* d_ws, size_t ws_size,
                              hipStream_t stream)
{
    const float* qvals  = (const float*)d_in[0];
    const float* states = (const float*)d_in[1];
    const float* l1a_w1 = (const float*)d_in[2];
    const float* l1a_b1 = (const float*)d_in[3];
    const float* l1a_w2 = (const float*)d_in[4];
    const float* l1a_b2 = (const float*)d_in[5];
    const float* l1e_w1 = (const float*)d_in[6];
    const float* l1e_b1 = (const float*)d_in[7];
    const float* l1e_w2 = (const float*)d_in[8];
    const float* l1e_b2 = (const float*)d_in[9];
    const float* l2a_w1 = (const float*)d_in[10];
    const float* l2a_b1 = (const float*)d_in[11];
    const float* l2a_w2 = (const float*)d_in[12];
    const float* l2a_b2 = (const float*)d_in[13];
    const float* l2e_w1 = (const float*)d_in[14];
    const float* l2e_b1 = (const float*)d_in[15];
    const float* l2e_w2 = (const float*)d_in[16];
    const float* l2e_b2 = (const float*)d_in[17];

    // Workspace layout: fp32 accumulators, then bf16-packed weight fragments.
    float* ws = (float*)d_ws;
    float* accA_hid = ws;                       // BT*64
    float* accE_w   = accA_hid + BT_N * EDIM;   // BT*64
    float* accE_b   = accE_w   + BT_N * EDIM;   // BT*64
    float* accA_w2  = accE_b   + BT_N * EDIM;   // BT*64
    float* accE_w2  = accA_w2  + BT_N * EDIM;   // BT*64
    float* accA_b2  = accE_w2  + BT_N * EDIM;   // BT
    float* accE_b2  = accA_b2  + BT_N;          // BT
    const size_t accFloats = (size_t)BT_N * EDIM * 5 + (size_t)BT_N * 2;

    unsigned int* pW1a = (unsigned int*)(ws + accFloats);   // 1*16*256 = 4096 dw
    unsigned int* pW2a = pW1a + 4096;                       // 8*8*256  = 16384 dw
    unsigned int* pW1c = pW2a + 16384;                      // 4096 dw
    unsigned int* pW2c = pW1c + 4096;                       // 8*5*256  = 10240 dw
    unsigned int* pW1e = pW2c + 10240;
    unsigned int* pW2e = pW1e + 4096;
    unsigned int* pW1f = pW2e + 16384;
    unsigned int* pW2f = pW1f + 4096;

    zero_acc<<<1024, 256, 0, stream>>>(ws, (int)accFloats);

    pack_w<<<64, 256, 0, stream>>>(l1a_w1, pW1a, 256, 1, 16);
    pack_w<<<64, 256, 0, stream>>>(l1a_w2, pW2a, 128, 8, 8);
    pack_w<<<64, 256, 0, stream>>>(l2a_w1, pW1c, 256, 1, 16);
    pack_w<<<64, 256, 0, stream>>>(l2a_w2, pW2c, 65,  8, 5);
    pack_w<<<64, 256, 0, stream>>>(l1e_w1, pW1e, 256, 1, 16);
    pack_w<<<64, 256, 0, stream>>>(l1e_w2, pW2e, 128, 8, 8);
    pack_w<<<64, 256, 0, stream>>>(l2e_w1, pW1f, 256, 1, 16);
    pack_w<<<64, 256, 0, stream>>>(l2e_w2, pW2f, 65,  8, 5);

    // Ally entities: BT*10 rows, 64 rows per block.
    entity_mlp<<<(BT_N * A_N) / 64, 128, 0, stream>>>(
        states, qvals,
        pW1a, l1a_b1, pW2a, l1a_b2,
        pW1c, l2a_b1, pW2c, l2a_b2,
        accA_hid, accA_hid, accA_w2, accA_b2,
        A_N, ALLY_OFF, 1);

    // Enemy entities: BT*11 rows.
    entity_mlp<<<(BT_N * NE_N) / 64, 128, 0, stream>>>(
        states, qvals,
        pW1e, l1e_b1, pW2e, l1e_b2,
        pW1f, l2e_b1, pW2f, l2e_b2,
        accE_w, accE_b, accE_w2, accE_b2,
        NE_N, ENEMY_OFF, 0);

    mixer<<<(BT_N + 255) / 256, 256, 0, stream>>>(
        qvals, accA_hid, accE_w, accE_b, accA_w2, accE_w2, accA_b2, accE_b2,
        (float*)d_out);
}
